// SparseMLHATransformerBlock_42356967473511
// MI455X (gfx1250) — matrix operand order repaired and checked by reading, not compiled, not run on hardware
//
#include <hip/hip_runtime.h>

typedef __bf16 bf16;
typedef __attribute__((ext_vector_type(16))) __bf16 v16bf;
typedef __attribute__((ext_vector_type(8)))  float  v8f;
typedef __attribute__((ext_vector_type(4)))  unsigned u32x4;
typedef __attribute__((ext_vector_type(2)))  unsigned u32x2;

#define S_   2048
#define D_   2048
#define H_   16
#define DN_  128
#define DR_  64
#define DV_  128
#define R_   512
#define HI_  4
#define DI_  64
#define KTOP_ 512
#define E_   4
#define DFF_ 8192
#define NEGF (-3.4028234663852886e38f)

// ---------------- helpers ----------------
__device__ __forceinline__ unsigned short f2bfbits(float f) {
  unsigned u = __builtin_bit_cast(unsigned, f);
  unsigned r = u + 0x7FFFu + ((u >> 16) & 1u);   // round-to-nearest-even
  return (unsigned short)(r >> 16);
}
__device__ __forceinline__ bf16 f2bf(float f) {
  unsigned short h = f2bfbits(f);
  return __builtin_bit_cast(bf16, h);
}

union Frag16 { v16bf v; unsigned u[8]; u32x4 q[2]; };

__device__ __forceinline__ v8f wmma_bf16(const v16bf& a, const v16bf& b, const v8f& c) {
  return __builtin_amdgcn_wmma_f32_16x16x32_bf16(false, a, false, b, (short)0, c, false, false);
}

// fp32 -> bf16 convert (weights)
__global__ void __launch_bounds__(256) cvt_bf16_kernel(const float* __restrict__ in,
                                                       bf16* __restrict__ out, long long n) {
  long long i  = (long long)blockIdx.x * blockDim.x + threadIdx.x;
  long long st = (long long)gridDim.x * blockDim.x;
  for (; i < n; i += st) out[i] = f2bf(in[i]);
}

// rmsnorm over `len` cols of a row, optional f32 and bf16 outputs
__global__ void __launch_bounds__(256) rmsnorm_kernel(const float* __restrict__ in, int instride, int len,
                                                      const float* __restrict__ w,
                                                      bf16* __restrict__ out_bf, int obstride,
                                                      float* __restrict__ out_f, int ofstride) {
  int row = blockIdx.x;
  const float* x = in + (size_t)row * instride;
  __shared__ float red[8];
  __shared__ float rsS;
  float s = 0.f;
  for (int i = threadIdx.x; i < len; i += 256) { float v = x[i]; s += v * v; }
  for (int off = 16; off; off >>= 1) s += __shfl_xor(s, off, 32);
  if ((threadIdx.x & 31) == 0) red[threadIdx.x >> 5] = s;
  __syncthreads();
  if (threadIdx.x == 0) {
    float tot = 0.f;
    for (int i = 0; i < 8; i++) tot += red[i];
    rsS = rsqrtf(tot / (float)len + 1e-6f);
  }
  __syncthreads();
  float rr = rsS;
  for (int i = threadIdx.x; i < len; i += 256) {
    float v = x[i] * rr * w[i];
    if (out_bf) out_bf[(size_t)row * obstride + i] = f2bf(v);
    if (out_f)  out_f[(size_t)row * ofstride + i] = v;
  }
}

// ---------------- bf16 WMMA GEMM, 128x128x32 tiles, 8 waves ----------------
// LDS layouts (dword units):
//   As_u[row*20 + k/2]                    row-major pairs, padded stride 20
//   Bs_u[col*20 + k/2]                    pair-interleaved col-major, stride 20
// Fragment loads are 2x b128, bank-conflict-free.
// EPI: 0 = f32 store; 1 = f32 + bias; 2 = bf16(gelu(x+bias)); 3 = f32 + resid
template <int EPI>
__global__ void __launch_bounds__(256)
gemm_bf16(const bf16* __restrict__ A, int lda,
          const bf16* __restrict__ Bw, int ldb,
          void* __restrict__ Cout, int ldc,
          int M, int N, int K,
          const int* __restrict__ rowmap,
          const int* __restrict__ cntPtr,
          const float* __restrict__ bias,
          const float* __restrict__ resid) {
  const int BM = 128, BN = 128, BK = 32;
  __shared__ __align__(16) unsigned As_u[BM * 20];
  __shared__ __align__(16) unsigned Bs_u[BN * 20];
  int m0 = blockIdx.y * BM, n0 = blockIdx.x * BN;
  int mEff = M;
  if (cntPtr) { int c = *cntPtr; mEff = (c < M) ? c : M; }
  if (m0 >= mEff) return;
  int t = threadIdx.x, lane = t & 31, wv = t >> 5;
  int wm = wv >> 2, wn = wv & 3;
  int half = lane >> 4, l15 = lane & 15;
  const v8f zv = {0.f, 0.f, 0.f, 0.f, 0.f, 0.f, 0.f, 0.f};
  const u32x4 z4 = {0u, 0u, 0u, 0u};
  v8f acc[4][2];
  for (int mt = 0; mt < 4; mt++) for (int nt = 0; nt < 2; nt++) acc[mt][nt] = zv;

  // prefetch registers
  u32x4 pa[2];
  u32x2 pb0[2], pb1[2];

  auto load_stage = [&](int k0) {
    #pragma unroll
    for (int i = 0; i < 2; i++) {
      int idx4 = i * 256 + t;              // 512 x b128 for A tile
      int row = idx4 >> 2, q4 = idx4 & 3;
      int gr = m0 + row;
      pa[i] = z4;
      if (gr < mEff) {
        int grow = rowmap ? rowmap[gr] : gr;
        pa[i] = ((const u32x4*)(A + (size_t)grow * lda + k0))[q4];
      }
    }
    #pragma unroll
    for (int i = 0; i < 2; i++) {
      int uid = i * 256 + t;               // 512 pack units (kp, 4 cols) for B tile
      int kp = uid >> 5, n4 = uid & 31;
      int gc = n0 + n4 * 4;
      int kk = k0 + kp * 2;
      u32x2 z2 = {0u, 0u};
      pb0[i] = z2; pb1[i] = z2;
      if (gc < N) {                        // N % 4 == 0 for all call sites
        const unsigned* bp = (const unsigned*)(Bw + (size_t)kk * ldb + gc);
        pb0[i] = *(const u32x2*)bp;
        pb1[i] = *(const u32x2*)(bp + (ldb >> 1));
      }
    }
  };
  auto store_stage = [&]() {
    u32x4* As4 = (u32x4*)As_u;
    #pragma unroll
    for (int i = 0; i < 2; i++) {
      int idx4 = i * 256 + t;
      int row = idx4 >> 2, q4 = idx4 & 3;
      As4[row * 5 + q4] = pa[i];
    }
    #pragma unroll
    for (int i = 0; i < 2; i++) {
      int uid = i * 256 + t;
      int kp = uid >> 5, n4 = uid & 31;
      unsigned u00 = pb0[i].x, u01 = pb0[i].y;
      unsigned u10 = pb1[i].x, u11 = pb1[i].y;
      unsigned base = (n4 * 4) * 20 + kp;
      Bs_u[base]      = (u00 & 0xFFFFu) | (u10 << 16);
      Bs_u[base + 20] = (u00 >> 16)     | (u10 & 0xFFFF0000u);
      Bs_u[base + 40] = (u01 & 0xFFFFu) | (u11 << 16);
      Bs_u[base + 60] = (u01 >> 16)     | (u11 & 0xFFFF0000u);
    }
  };

  load_stage(0);
  int nk = K / BK;
  for (int kb = 0; kb < nk; kb++) {
    __syncthreads();
    store_stage();
    __syncthreads();
    if (kb + 1 < nk) load_stage((kb + 1) * BK);

    Frag16 af[4], bfr[2];
    const u32x4* As4 = (const u32x4*)As_u;
    const u32x4* Bs4 = (const u32x4*)Bs_u;
    #pragma unroll
    for (int mt = 0; mt < 4; mt++) {
      int row = wm * 64 + mt * 16 + l15;
      af[mt].q[0] = As4[row * 5 + half];
      af[mt].q[1] = As4[row * 5 + 2 + half];
    }
    #pragma unroll
    for (int nt = 0; nt < 2; nt++) {
      int col = wn * 32 + nt * 16 + l15;
      bfr[nt].q[0] = Bs4[col * 5 + half];
      bfr[nt].q[1] = Bs4[col * 5 + 2 + half];
    }
    #pragma unroll
    for (int mt = 0; mt < 4; mt++)
      #pragma unroll
      for (int nt = 0; nt < 2; nt++)
        acc[mt][nt] = wmma_bf16(af[mt].v, bfr[nt].v, acc[mt][nt]);
  }

  // epilogue (C layout: lane -> col l15, vgpr r -> row r + 8*half)
  #pragma unroll
  for (int mt = 0; mt < 4; mt++) {
    #pragma unroll
    for (int nt = 0; nt < 2; nt++) {
      #pragma unroll
      for (int r = 0; r < 8; r++) {
        int row = m0 + wm * 64 + mt * 16 + r + 8 * half;
        int col = n0 + wn * 32 + nt * 16 + l15;
        if (row < mEff && col < N) {
          float v = acc[mt][nt][r];
          if (EPI == 1) {
            ((float*)Cout)[(size_t)row * ldc + col] = v + bias[col];
          } else if (EPI == 2) {
            float g = v + bias[col];
            g = 0.5f * g * (1.0f + erff(g * 0.70710678118654752f));
            ((bf16*)Cout)[(size_t)row * ldc + col] = f2bf(g);
          } else if (EPI == 3) {
            ((float*)Cout)[(size_t)row * ldc + col] = v + resid[(size_t)row * ldc + col];
          } else {
            ((float*)Cout)[(size_t)row * ldc + col] = v;
          }
        }
      }
    }
  }
}

// ---------------- rope + head-major relayout ----------------
__global__ void __launch_bounds__(64)
prep_qkv_kernel(const float* __restrict__ q, const float* __restrict__ kv,
                const float* __restrict__ ckv,
                bf16* __restrict__ qfb, bf16* __restrict__ kfb, bf16* __restrict__ vfb) {
  int s = blockIdx.x, h = blockIdx.y, t = threadIdx.x;
  size_t base = ((size_t)h * S_ + s) * 192;
  const float* qrow  = q  + (size_t)s * (H_ * (DN_ + DR_)) + h * (DN_ + DR_);
  const float* kvrow = kv + (size_t)s * (H_ * (DN_ + DV_)) + h * (DN_ + DV_);
  for (int i = t; i < 128; i += 64) {
    qfb[base + i] = f2bf(qrow[i]);
    kfb[base + i] = f2bf(kvrow[i]);
    vfb[((size_t)h * S_ + s) * 128 + i] = f2bf(kvrow[128 + i]);
  }
  if (t < 32) {
    float fr = powf(10000.f, -(float)t / 32.f);
    float ang = (float)s * fr;
    float cs = cosf(ang), sn = sinf(ang);
    float x1 = qrow[128 + t], x2 = qrow[128 + 32 + t];
    qfb[base + 128 + t]      = f2bf(x1 * cs - x2 * sn);
    qfb[base + 128 + 32 + t] = f2bf(x2 * cs + x1 * sn);
    float k1 = ckv[(size_t)s * (R_ + DR_) + R_ + t];
    float k2 = ckv[(size_t)s * (R_ + DR_) + R_ + 32 + t];
    kfb[base + 128 + t]      = f2bf(k1 * cs - k2 * sn);
    kfb[base + 128 + 32 + t] = f2bf(k2 * cs + k1 * sn);
  }
}

// ---------------- indexer scores (fp32) ----------------
__global__ void __launch_bounds__(256)
idx_scores_kernel(const float* __restrict__ qi, const float* __restrict__ ki,
                  const float* __restrict__ wgt, float* __restrict__ idx_out) {
  int q0 = blockIdx.y * 16, k0 = blockIdx.x * 16;
  __shared__ float Kis[16 * 68];   // padded stride: conflict-free
  __shared__ float Qis[16 * 256];
  __shared__ float Wsh[16 * 4];
  int t = threadIdx.x;
  for (int i = t; i < 16 * 64; i += 256) Kis[(i >> 6) * 68 + (i & 63)] = ki[(size_t)(k0 + (i >> 6)) * 64 + (i & 63)];
  for (int i = t; i < 16 * 256; i += 256) Qis[i] = qi[(size_t)q0 * 256 + i];
  if (t < 64) Wsh[t] = wgt[(size_t)q0 * 4 + t];
  __syncthreads();
  int tq = t >> 4, tk = t & 15;
  int qq = q0 + tq, kk = k0 + tk;
  float out = NEGF;
  if (kk <= qq) {
    float s = 0.f;
    #pragma unroll
    for (int hh = 0; hh < 4; hh++) {
      float d = 0.f;
      #pragma unroll
      for (int j = 0; j < 64; j++) d += Qis[tq * 256 + hh * 64 + j] * Kis[tk * 68 + j];
      s += fmaxf(d, 0.f) * Wsh[tq * 4 + hh];
    }
    out = s;
  }
  idx_out[(size_t)qq * S_ + kk] = out;
}

// ---------------- per-row top-512 radix select -> key bitmask ----------------
__global__ void __launch_bounds__(256)
topk_mask_kernel(const float* __restrict__ idx, unsigned* __restrict__ maskbits) {
  int q = blockIdx.x;
  __shared__ unsigned Ls[S_];
  __shared__ int cntS;
  int t = threadIdx.x;
  int n = q + 1;
  for (int i = t; i < S_; i += 256) {
    unsigned u = 0u;
    if (i < n) {
      unsigned b = __builtin_bit_cast(unsigned, idx[(size_t)q * S_ + i]);
      u = (b & 0x80000000u) ? ~b : (b | 0x80000000u);
    }
    Ls[i] = u;
  }
  __syncthreads();
  unsigned T = 0u;
  if (n > KTOP_) {
    int kk = KTOP_;
    unsigned prefix = 0u;
    for (int b = 31; b >= 0; b--) {
      if (t == 0) cntS = 0;
      __syncthreads();
      unsigned hiMask = (b == 31) ? 0u : (0xFFFFFFFFu << (b + 1));
      int local = 0;
      for (int i = t; i < n; i += 256) {
        unsigned u = Ls[i];
        if (((u & hiMask) == prefix) && ((u >> b) & 1u)) local++;
      }
      atomicAdd(&cntS, local);
      __syncthreads();
      int c = cntS;
      if (c >= kk) prefix |= (1u << b); else kk -= c;
      __syncthreads();
    }
    T = prefix;
  }
  for (int w = t; w < S_ / 32; w += 256) {
    unsigned word = 0u;
    #pragma unroll
    for (int j = 0; j < 32; j++) {
      int key = w * 32 + j;
      if (key < n && Ls[key] >= T) word |= (1u << j);
    }
    maskbits[(size_t)q * (S_ / 32) + w] = word;
  }
}

// ---------------- flash attention (masked, bf16 WMMA) ----------------
// LDS: Ks row-major [32][200] bf16 (stride 100 dwords, padded),
//      Vs pair-interleaved col-major Vs_u[col*20 + kpair],
//      Ps per-wave [16][40] ushort (stride 20 dwords).
__global__ void __launch_bounds__(256)
attn_kernel(const bf16* __restrict__ qf, const bf16* __restrict__ kf,
            const bf16* __restrict__ vf, const unsigned* __restrict__ maskbits,
            bf16* __restrict__ obuf) {
  const int h = blockIdx.y;
  const int qb0 = blockIdx.x * 128;
  __shared__ __align__(16) unsigned Ks_u[32 * 100];
  __shared__ __align__(16) unsigned Vs_u[128 * 20];
  __shared__ __align__(16) unsigned short Ps[8][640];
  int t = threadIdx.x, lane = t & 31, wv = t >> 5;
  int half = lane >> 4, l15 = lane & 15;
  int qrow = qb0 + wv * 16 + l15;

  Frag16 aq[6];
  const u32x4* q4p = (const u32x4*)(qf + ((size_t)h * S_ + qrow) * 192);
  #pragma unroll
  for (int ck = 0; ck < 6; ck++) {
    aq[ck].q[0] = q4p[ck * 4 + half];
    aq[ck].q[1] = q4p[ck * 4 + 2 + half];
  }

  const v8f zv = {0.f, 0.f, 0.f, 0.f, 0.f, 0.f, 0.f, 0.f};
  v8f oacc[8];
  float mrow[8], lrow[8];
  #pragma unroll
  for (int i = 0; i < 8; i++) { oacc[i] = zv; mrow[i] = -1e30f; lrow[i] = 0.f; }

  // staging prefetch registers
  u32x4 kreg[3];
  u32x2 vr0[2], vr1[2];
  auto load_kv = [&](int kb) {
    int key0 = kb * 32;
    #pragma unroll
    for (int i = 0; i < 3; i++) {
      int idx4 = i * 256 + t;                  // 768 x b128 for K tile (32 rows x 24 b128)
      int row = idx4 / 24, c4 = idx4 % 24;
      kreg[i] = ((const u32x4*)(kf + ((size_t)h * S_ + key0 + row) * 192))[c4];
    }
    #pragma unroll
    for (int i = 0; i < 2; i++) {
      int uid = i * 256 + t;                   // 512 pack units for V tile
      int kp = uid >> 5, n4 = uid & 31;
      const unsigned* vp = (const unsigned*)(vf + ((size_t)h * S_ + key0 + kp * 2) * 128 + n4 * 4);
      vr0[i] = *(const u32x2*)vp;
      vr1[i] = *(const u32x2*)(vp + 64);       // next key row (+128 bf16)
    }
  };
  auto store_kv = [&]() {
    u32x4* Ks4 = (u32x4*)Ks_u;
    #pragma unroll
    for (int i = 0; i < 3; i++) {
      int idx4 = i * 256 + t;
      int row = idx4 / 24, c4 = idx4 % 24;
      Ks4[row * 25 + c4] = kreg[i];
    }
    #pragma unroll
    for (int i = 0; i < 2; i++) {
      int uid = i * 256 + t;
      int kp = uid >> 5, n4 = uid & 31;
      unsigned u00 = vr0[i].x, u01 = vr0[i].y;
      unsigned u10 = vr1[i].x, u11 = vr1[i].y;
      unsigned base = (n4 * 4) * 20 + kp;
      Vs_u[base]      = (u00 & 0xFFFFu) | (u10 << 16);
      Vs_u[base + 20] = (u00 >> 16)     | (u10 & 0xFFFF0000u);
      Vs_u[base + 40] = (u01 & 0xFFFFu) | (u11 << 16);
      Vs_u[base + 60] = (u01 >> 16)     | (u11 & 0xFFFF0000u);
    }
  };

  const float scale = 0.07216878364870322f; // 1/sqrt(192)
  int nkb = (qb0 + 128) / 32;
  load_kv(0);
  for (int kb = 0; kb < nkb; kb++) {
    __syncthreads();
    store_kv();
    __syncthreads();
    if (kb + 1 < nkb) load_kv(kb + 1);

    // scores: 16x32 (two 16x16 tiles), K=192 over 6 chunks
    v8f c0 = zv, c1 = zv;
    const u32x4* Ks4 = (const u32x4*)Ks_u;
    #pragma unroll
    for (int ck = 0; ck < 6; ck++) {
      Frag16 b0, b1;
      b0.q[0] = Ks4[l15 * 25 + ck * 4 + half];
      b0.q[1] = Ks4[l15 * 25 + ck * 4 + 2 + half];
      b1.q[0] = Ks4[(16 + l15) * 25 + ck * 4 + half];
      b1.q[1] = Ks4[(16 + l15) * 25 + ck * 4 + 2 + half];
      c0 = wmma_bf16(aq[ck].v, b0.v, c0);
      c1 = wmma_bf16(aq[ck].v, b1.v, c1);
    }

    unsigned myw = 0u;
    if (lane < 16) myw = maskbits[(size_t)(qb0 + wv * 16 + lane) * (S_ / 32) + kb];

    float pr0[8], pr1[8];
    #pragma unroll
    for (int r = 0; r < 8; r++) {
      unsigned w = __shfl(myw, r + 8 * half, 32);
      bool m0 = (w >> l15) & 1u;
      bool m1 = (w >> (16 + l15)) & 1u;
      float s0 = m0 ? c0[r] * scale : -1e30f;
      float s1 = m1 ? c1[r] * scale : -1e30f;
      float mx = fmaxf(s0, s1);
      for (int off = 1; off < 16; off <<= 1) mx = fmaxf(mx, __shfl_xor(mx, off, 32));
      float newm = fmaxf(mrow[r], mx);
      float cf = __expf(mrow[r] - newm);
      float p0 = m0 ? __expf(s0 - newm) : 0.f;
      float p1 = m1 ? __expf(s1 - newm) : 0.f;
      float ps = p0 + p1;
      for (int off = 1; off < 16; off <<= 1) ps += __shfl_xor(ps, off, 32);
      lrow[r] = lrow[r] * cf + ps;
      mrow[r] = newm;
      pr0[r] = p0; pr1[r] = p1;
      #pragma unroll
      for (int nt = 0; nt < 8; nt++) oacc[nt][r] *= cf;
    }

    // P tile C-layout -> A-layout via per-wave LDS (stride 40 ushorts)
    unsigned short* pp = &Ps[wv][0];
    #pragma unroll
    for (int r = 0; r < 8; r++) {
      int row = r + 8 * half;
      pp[row * 40 + l15]      = f2bfbits(pr0[r]);
      pp[row * 40 + 16 + l15] = f2bfbits(pr1[r]);
    }
    asm volatile("s_wait_dscnt 0" ::: "memory");
    __builtin_amdgcn_wave_barrier();
    Frag16 pa;
    {
      const u32x4* Pp4 = (const u32x4*)&Ps[wv][0];
      pa.q[0] = Pp4[l15 * 5 + half];
      pa.q[1] = Pp4[l15 * 5 + 2 + half];
    }
    // o += P(16x32) @ V(32x128)
    const u32x4* Vs4 = (const u32x4*)Vs_u;
    #pragma unroll
    for (int nt = 0; nt < 8; nt++) {
      Frag16 bv;
      int col = nt * 16 + l15;
      bv.q[0] = Vs4[col * 5 + half];
      bv.q[1] = Vs4[col * 5 + 2 + half];
      oacc[nt] = wmma_bf16(pa.v, bv.v, oacc[nt]);
    }
  }

  #pragma unroll
  for (int r = 0; r < 8; r++) {
    int row = qb0 + wv * 16 + r + 8 * half;
    float inv = (lrow[r] > 0.f) ? (1.0f / lrow[r]) : 0.f;
    #pragma unroll
    for (int nt = 0; nt < 8; nt++)
      obuf[(size_t)row * (H_ * DV_) + h * DV_ + nt * 16 + l15] = f2bf(oacc[nt][r] * inv);
  }
}

// ---------------- small N<=8 matvec per row (wgt / router logits) ----------------
__global__ void __launch_bounds__(256)
smallmat_kernel(const float* __restrict__ in, int lda,
                const float* __restrict__ W, int N, float* __restrict__ out) {
  int s = blockIdx.x;
  float acc[8];
  for (int n = 0; n < 8; n++) acc[n] = 0.f;
  for (int i = threadIdx.x; i < D_; i += 256) {
    float v = in[(size_t)s * lda + i];
    for (int n = 0; n < N; n++) acc[n] += v * W[(size_t)i * N + n];
  }
  for (int n = 0; n < N; n++)
    for (int off = 16; off; off >>= 1) acc[n] += __shfl_xor(acc[n], off, 32);
  __shared__ float red[8][8];
  int wv = threadIdx.x >> 5, lane = threadIdx.x & 31;
  if (lane == 0) for (int n = 0; n < N; n++) red[wv][n] = acc[n];
  __syncthreads();
  if (threadIdx.x == 0)
    for (int n = 0; n < N; n++) {
      float tot = 0.f;
      for (int w = 0; w < 8; w++) tot += red[w][n];
      out[(size_t)s * N + n] = tot;
    }
}

// ---------------- router: softmax-4, top-2, expert grouping, aux sums ----------------
__global__ void __launch_bounds__(256)
router_kernel(const float* __restrict__ logits, int* __restrict__ counts,
              int* __restrict__ rowmap, int* __restrict__ slots,
              float* __restrict__ gates, float* __restrict__ sumP) {
  int s = blockIdx.x * blockDim.x + threadIdx.x;
  if (s >= S_) return;
  float l[4], p[4];
  float mx = -1e30f;
  for (int e = 0; e < 4; e++) { l[e] = logits[(size_t)s * 4 + e]; mx = fmaxf(mx, l[e]); }
  float sum = 0.f;
  for (int e = 0; e < 4; e++) { p[e] = __expf(l[e] - mx); sum += p[e]; }
  for (int e = 0; e < 4; e++) p[e] /= sum;
  int i0 = 0;
  for (int e = 1; e < 4; e++) if (p[e] > p[i0]) i0 = e;
  int i1 = -1;
  for (int e = 0; e < 4; e++) if (e != i0 && (i1 < 0 || p[e] > p[i1])) i1 = e;
  float g = p[i0] + p[i1];
  int ids[2] = {i0, i1};
  float gv[2] = {p[i0] / g, p[i1] / g};
  for (int j = 0; j < 2; j++) {
    int e = ids[j];
    int pos = atomicAdd(&counts[e], 1);
    rowmap[e * S_ + pos] = s;
    slots[s * 2 + j] = e * S_ + pos;
    gates[s * 2 + j] = gv[j];
  }
  for (int e = 0; e < 4; e++) atomicAdd(&sumP[e], p[e]);
}

__global__ void aux_kernel(const int* __restrict__ counts, const float* __restrict__ sumP,
                           float* __restrict__ out) {
  if (threadIdx.x == 0 && blockIdx.x == 0) {
    float a = 0.f;
    for (int e = 0; e < 4; e++)
      a += ((float)counts[e] / (float)S_) * (sumP[e] / (float)S_);
    out[0] = (float)E_ * a;
  }
}

// ---------------- gate-weighted expert combine into x ----------------
__global__ void __launch_bounds__(256)
moe_combine_kernel(const float* __restrict__ eout, const int* __restrict__ slots,
                   const float* __restrict__ gates, float* __restrict__ x) {
  int s = blockIdx.x;
  int s0 = slots[s * 2], s1 = slots[s * 2 + 1];
  float g0 = gates[s * 2], g1 = gates[s * 2 + 1];
  for (int i = threadIdx.x; i < D_; i += 256)
    x[(size_t)s * D_ + i] += g0 * eout[(size_t)s0 * D_ + i] + g1 * eout[(size_t)s1 * D_ + i];
}

// ================= host =================
extern "C" void kernel_launch(void* const* d_in, const int* in_sizes, int n_in,
                              void* d_out, int out_size, void* d_ws, size_t ws_size,
                              hipStream_t stream) {
  const float* x    = (const float*)d_in[0];
  const float* n1w  = (const float*)d_in[1];
  const float* n2w  = (const float*)d_in[2];
  const float* kvnw = (const float*)d_in[3];
  const float* Wq   = (const float*)d_in[4];
  const float* Wkva = (const float*)d_in[5];
  const float* Wkvb = (const float*)d_in[6];
  const float* Wo   = (const float*)d_in[7];
  const float* Wiq  = (const float*)d_in[8];
  const float* Wik  = (const float*)d_in[9];
  const float* Wiw  = (const float*)d_in[10];
  const float* Wr   = (const float*)d_in[11];
  const float* W1   = (const float*)d_in[12];
  const float* b1   = (const float*)d_in[13];
  const float* W2   = (const float*)d_in[14];
  const float* b2   = (const float*)d_in[15];

  float* xo   = (float*)d_out;                 // S*D
  float* auxo = xo + (size_t)S_ * D_;          // 1
  float* idxo = auxo + 1;                      // S*S

  char* wsp = (char*)d_ws;
  size_t off = 0;
  auto alloc = [&](size_t bytes) -> char* {
    char* p = wsp + off;
    off += (bytes + 255) & ~(size_t)255;
    return p;
  };

  bf16*  h_bf    = (bf16*)alloc((size_t)S_ * D_ * 2);
  float* h_f     = (float*)alloc((size_t)S_ * D_ * 4);
  bf16*  Wq_bf   = (bf16*)alloc((size_t)D_ * 3072 * 2);
  bf16*  Wkva_bf = (bf16*)alloc((size_t)D_ * 576 * 2);
  bf16*  Wkvb_bf = (bf16*)alloc((size_t)R_ * 4096 * 2);
  bf16*  Wo_bf   = (bf16*)alloc((size_t)D_ * D_ * 2);
  bf16*  Wiq_bf  = (bf16*)alloc((size_t)D_ * 256 * 2);
  bf16*  Wik_bf  = (bf16*)alloc((size_t)D_ * 64 * 2);
  bf16*  W1_bf   = (bf16*)alloc((size_t)E_ * D_ * DFF_ * 2);
  bf16*  W2_bf   = (bf16*)alloc((size_t)E_ * DFF_ * D_ * 2);
  float* qbuf    = (float*)alloc((size_t)S_ * 3072 * 4);
  float* ckv     = (float*)alloc((size_t)S_ * 576 * 4);
  bf16*  c_bf    = (bf16*)alloc((size_t)S_ * R_ * 2);
  float* kvbuf   = (float*)alloc((size_t)S_ * 4096 * 4);
  bf16*  qfb     = (bf16*)alloc((size_t)H_ * S_ * 192 * 2);
  bf16*  kfb     = (bf16*)alloc((size_t)H_ * S_ * 192 * 2);
  bf16*  vfb     = (bf16*)alloc((size_t)H_ * S_ * 128 * 2);
  float* qi      = (float*)alloc((size_t)S_ * 256 * 4);
  float* ki      = (float*)alloc((size_t)S_ * 64 * 4);
  float* wgt     = (float*)alloc((size_t)S_ * 4 * 4);
  unsigned* maskb= (unsigned*)alloc((size_t)S_ * (S_ / 32) * 4);
  bf16*  obuf    = (bf16*)alloc((size_t)S_ * D_ * 2);
  float* h2f     = (float*)alloc((size_t)S_ * D_ * 4);
  bf16*  h2_bf   = (bf16*)alloc((size_t)S_ * D_ * 2);
  float* rlog    = (float*)alloc((size_t)S_ * 4 * 4);
  int*   counts  = (int*)alloc(4 * 4 + 4 * 4);
  float* sumP    = (float*)(counts + 4);
  int*   rowmap  = (int*)alloc((size_t)E_ * S_ * 4);
  int*   slots   = (int*)alloc((size_t)2 * S_ * 4);
  float* gates   = (float*)alloc((size_t)2 * S_ * 4);
  bf16*  hid_bf  = (bf16*)alloc((size_t)E_ * S_ * DFF_ * 2);
  float* eout    = (float*)alloc((size_t)E_ * S_ * D_ * 4);

  // 1. weight conversion
  auto cvt = [&](const float* src, bf16* dst, long long n) {
    cvt_bf16_kernel<<<4096, 256, 0, stream>>>(src, dst, n);
  };
  cvt(Wq, Wq_bf, (long long)D_ * 3072);
  cvt(Wkva, Wkva_bf, (long long)D_ * 576);
  cvt(Wkvb, Wkvb_bf, (long long)R_ * 4096);
  cvt(Wo, Wo_bf, (long long)D_ * D_);
  cvt(Wiq, Wiq_bf, (long long)D_ * 256);
  cvt(Wik, Wik_bf, (long long)D_ * 64);
  cvt(W1, W1_bf, (long long)E_ * D_ * DFF_);
  cvt(W2, W2_bf, (long long)E_ * DFF_ * D_);

  // 2. h = rms(x, norm1)
  rmsnorm_kernel<<<S_, 256, 0, stream>>>(x, D_, D_, n1w, h_bf, D_, h_f, D_);

  // 3. projections
  gemm_bf16<0><<<dim3(24, 16), 256, 0, stream>>>(h_bf, D_, Wq_bf, 3072, qbuf, 3072,
                                                 S_, 3072, D_, nullptr, nullptr, nullptr, nullptr);
  gemm_bf16<0><<<dim3(5, 16), 256, 0, stream>>>(h_bf, D_, Wkva_bf, 576, ckv, 576,
                                                S_, 576, D_, nullptr, nullptr, nullptr, nullptr);
  // 4. c = rms(ckv[:,:512], kv_norm)
  rmsnorm_kernel<<<S_, 256, 0, stream>>>(ckv, 576, R_, kvnw, c_bf, R_, nullptr, 0);
  // 5. kv = c @ Wkvb
  gemm_bf16<0><<<dim3(32, 16), 256, 0, stream>>>(c_bf, R_, Wkvb_bf, 4096, kvbuf, 4096,
                                                 S_, 4096, R_, nullptr, nullptr, nullptr, nullptr);
  // 6. indexer projections
  gemm_bf16<0><<<dim3(2, 16), 256, 0, stream>>>(h_bf, D_, Wiq_bf, 256, qi, 256,
                                                S_, 256, D_, nullptr, nullptr, nullptr, nullptr);
  gemm_bf16<0><<<dim3(1, 16), 256, 0, stream>>>(h_bf, D_, Wik_bf, 64, ki, 64,
                                                S_, 64, D_, nullptr, nullptr, nullptr, nullptr);
  smallmat_kernel<<<S_, 256, 0, stream>>>(h_f, D_, Wiw, HI_, wgt);

  // 7. rope + head-major layouts
  prep_qkv_kernel<<<dim3(S_, H_), 64, 0, stream>>>(qbuf, kvbuf, ckv, qfb, kfb, vfb);

  // 8. indexer scores -> d_out idx region; per-row top-512 -> bitmask
  idx_scores_kernel<<<dim3(128, 128), 256, 0, stream>>>(qi, ki, wgt, idxo);
  topk_mask_kernel<<<S_, 256, 0, stream>>>(idxo, maskb);

  // 9. masked flash attention
  attn_kernel<<<dim3(16, 16), 256, 0, stream>>>(qfb, kfb, vfb, maskb, obuf);

  // 10. x_mid = x + o @ Wo  (written into d_out x region)
  gemm_bf16<3><<<dim3(16, 16), 256, 0, stream>>>(obuf, D_, Wo_bf, D_, xo, D_,
                                                 S_, D_, D_, nullptr, nullptr, nullptr, x);

  // 11. h2 = rms(x_mid, norm2)
  rmsnorm_kernel<<<S_, 256, 0, stream>>>(xo, D_, D_, n2w, h2_bf, D_, h2f, D_);

  // 12. router
  smallmat_kernel<<<S_, 256, 0, stream>>>(h2f, D_, Wr, E_, rlog);
  hipMemsetAsync(counts, 0, 4 * 4 + 4 * 4, stream);
  router_kernel<<<8, 256, 0, stream>>>(rlog, counts, rowmap, slots, gates, sumP);
  aux_kernel<<<1, 32, 0, stream>>>(counts, sumP, auxo);

  // 13. grouped expert FFN (only selected tokens per expert)
  for (int e = 0; e < E_; e++) {
    gemm_bf16<2><<<dim3(64, 16), 256, 0, stream>>>(
        h2_bf, D_, W1_bf + (size_t)e * D_ * DFF_, DFF_,
        hid_bf + (size_t)e * S_ * DFF_, DFF_,
        S_, DFF_, D_, rowmap + e * S_, counts + e, b1 + (size_t)e * DFF_, nullptr);
    gemm_bf16<1><<<dim3(16, 16), 256, 0, stream>>>(
        hid_bf + (size_t)e * S_ * DFF_, DFF_, W2_bf + (size_t)e * DFF_ * D_, D_,
        eout + (size_t)e * S_ * D_, D_,
        S_, D_, DFF_, nullptr, counts + e, b2 + (size_t)e * D_, nullptr);
  }

  // 14. x_out = x_mid + gate-weighted expert outputs
  moe_combine_kernel<<<S_, 256, 0, stream>>>(eout, slots, gates, xo);

  (void)in_sizes; (void)n_in; (void)out_size; (void)ws_size;
}
